// Decoder_89292370084325
// MI455X (gfx1250) — compile-verified, matrix operand order checked by reading, not run on hardware
//
#include <hip/hip_runtime.h>
#include <hip/hip_bf16.h>
#include <stdint.h>
#include <stddef.h>

// Problem dims (match reference)
#define BB 64
#define TT 19
#define SS 80
#define VV 32000
#define EE 256
#define UU 1024

typedef __attribute__((ext_vector_type(16))) __bf16 v16bf;
typedef __attribute__((ext_vector_type(8)))  __bf16 v8bf;
typedef __attribute__((ext_vector_type(8)))  float  v8f;

// ---------------------------------------------------------------------------
// f32 -> bf16 elementwise convert
// ---------------------------------------------------------------------------
__global__ void k_f32_to_bf16(const float* __restrict__ in, __bf16* __restrict__ out, int n) {
    int i = blockIdx.x * blockDim.x + threadIdx.x;
    if (i < n) out[i] = (__bf16)in[i];
}

// ---------------------------------------------------------------------------
// f32 [R,C] row-major  ->  bf16 [C,R] row-major (transpose), LDS-tiled
// ---------------------------------------------------------------------------
__global__ void k_transpose_f32_to_bf16(const float* __restrict__ in, __bf16* __restrict__ out,
                                        int R, int C) {
    __shared__ float tile[32][33];
    int c0 = blockIdx.x * 32, r0 = blockIdx.y * 32;
    int tx = threadIdx.x, ty = threadIdx.y;        // blockDim = (32, 8)
    #pragma unroll
    for (int i = ty; i < 32; i += 8) {
        int r = r0 + i, c = c0 + tx;
        tile[i][tx] = (r < R && c < C) ? in[(size_t)r * C + c] : 0.f;
    }
    __syncthreads();
    #pragma unroll
    for (int i = ty; i < 32; i += 8) {
        int c = c0 + i, r = r0 + tx;               // out[c][r]
        if (c < C && r < R) out[(size_t)c * R + r] = (__bf16)tile[tx][i];
    }
}

// ---------------------------------------------------------------------------
// WMMA bf16 GEMM:  C[M,N] (f32) = A[M,K] (bf16, row-major) * Bt[N,K]^T (bf16)
// Requirements here: M % 64 == 0, N % 128 == 0, K % 32 == 0.
// One wave owns a 16-wide N strip and 4 M-tiles (64 rows) => B-frag reused 4x.
// Fragment layout per CDNA5 ISA (16-bit A/B 16x32):
//   lane l: row/col = l&15, khalf = l>>4; elems 0..7 -> k = khalf*8 + 0..7,
//   elems 8..15 -> k = khalf*8 + 16..23  => two 16B contiguous loads.
// ---------------------------------------------------------------------------
__device__ __forceinline__ v16bf load_frag(const __bf16* p) {
    v8bf lo = *(const v8bf*)(p);
    v8bf hi = *(const v8bf*)(p + 16);
    return __builtin_shufflevector(lo, hi, 0,1,2,3,4,5,6,7,8,9,10,11,12,13,14,15);
}

__global__ __launch_bounds__(256)
void k_wmma_gemm(const __bf16* __restrict__ A, const __bf16* __restrict__ Bt,
                 float* __restrict__ C, const float* __restrict__ bias,
                 int M, int N, int K, int ldc) {
    int w    = threadIdx.x >> 5;
    int lane = threadIdx.x & 31;
    int tn   = blockIdx.x * 8 + w;          // 16-wide N tile index
    int m0   = blockIdx.y * 64;             // 64 rows of M per block row
    if (tn * 16 >= N) return;               // wave-uniform

    int half = lane >> 4;                   // k-half select
    int rc   = lane & 15;                   // row (A) / col (B)

    const __bf16* bp  = Bt + (size_t)(tn * 16 + rc) * K + half * 8;
    const __bf16* ap0 = A  + (size_t)(m0 + rc) * K + half * 8;

    v8f acc0 = {}, acc1 = {}, acc2 = {}, acc3 = {};
    for (int k = 0; k < K; k += 32) {
        v16bf bf = load_frag(bp + k);
        v16bf a0 = load_frag(ap0 + k);
        v16bf a1 = load_frag(ap0 + (size_t)16 * K + k);
        v16bf a2 = load_frag(ap0 + (size_t)32 * K + k);
        v16bf a3 = load_frag(ap0 + (size_t)48 * K + k);
        acc0 = __builtin_amdgcn_wmma_f32_16x16x32_bf16(false, a0, false, bf, (short)0, acc0, false, false);
        acc1 = __builtin_amdgcn_wmma_f32_16x16x32_bf16(false, a1, false, bf, (short)0, acc1, false, false);
        acc2 = __builtin_amdgcn_wmma_f32_16x16x32_bf16(false, a2, false, bf, (short)0, acc2, false, false);
        acc3 = __builtin_amdgcn_wmma_f32_16x16x32_bf16(false, a3, false, bf, (short)0, acc3, false, false);
    }

    int n  = tn * 16 + rc;
    float bv = bias ? bias[n] : 0.f;
    int mb = m0 + half * 8;                 // D layout: VGPR j -> M = j + 8*(lane>=16)
    #pragma unroll
    for (int j = 0; j < 8; ++j) C[(size_t)(mb      + j) * ldc + n] = acc0[j] + bv;
    #pragma unroll
    for (int j = 0; j < 8; ++j) C[(size_t)(mb + 16 + j) * ldc + n] = acc1[j] + bv;
    #pragma unroll
    for (int j = 0; j < 8; ++j) C[(size_t)(mb + 32 + j) * ldc + n] = acc2[j] + bv;
    #pragma unroll
    for (int j = 0; j < 8; ++j) C[(size_t)(mb + 48 + j) * ldc + n] = acc3[j] + bv;
}

// ---------------------------------------------------------------------------
// Init: h = enc_state (f32 + bf16 copy), attn_prev = 0
// ---------------------------------------------------------------------------
__global__ void k_init(const float* __restrict__ enc, float* __restrict__ h,
                       __bf16* __restrict__ hB, float* __restrict__ attnF) {
    int i = blockIdx.x * blockDim.x + threadIdx.x;   // BB*UU
    if (i >= BB * UU) return;
    float v = enc[i];
    h[i] = v; hB[i] = (__bf16)v; attnF[i] = 0.f;
}

// ---------------------------------------------------------------------------
// cell_in = [ embedding[x[:,t]] | attn_prev ]  -> bf16 [BB, EE+UU]
// ---------------------------------------------------------------------------
__global__ void k_build_cellin(const int* __restrict__ x, const float* __restrict__ emb,
                               const float* __restrict__ attnF, __bf16* __restrict__ cellin,
                               int t) {
    int i = blockIdx.x * blockDim.x + threadIdx.x;   // BB*(EE+UU)
    if (i >= BB * (EE + UU)) return;
    int b = i / (EE + UU), c = i % (EE + UU);
    float v;
    if (c < EE) v = emb[(size_t)x[b * TT + t] * EE + c];
    else        v = attnF[b * UU + (c - EE)];
    cellin[i] = (__bf16)v;
}

// ---------------------------------------------------------------------------
// Keras GRU (reset_after=True, gate order [z,r,h]):  h <- update in place
// ---------------------------------------------------------------------------
__global__ void k_gru_gate(const float* __restrict__ xm, const float* __restrict__ hm,
                           const float* __restrict__ gbias, float* __restrict__ h,
                           __bf16* __restrict__ hB) {
    int i = blockIdx.x * blockDim.x + threadIdx.x;   // BB*UU
    if (i >= BB * UU) return;
    int b = i >> 10, u = i & (UU - 1);
    const float* bin = gbias;
    const float* brc = gbias + 3 * UU;
    size_t r0 = (size_t)b * 3 * UU;
    float xz = xm[r0 + u]          + bin[u];
    float xr = xm[r0 + UU + u]     + bin[UU + u];
    float xh = xm[r0 + 2 * UU + u] + bin[2 * UU + u];
    float hz = hm[r0 + u]          + brc[u];
    float hr = hm[r0 + UU + u]     + brc[UU + u];
    float hhm = hm[r0 + 2 * UU + u] + brc[2 * UU + u];
    float z  = 1.f / (1.f + __expf(-(xz + hz)));
    float r  = 1.f / (1.f + __expf(-(xr + hr)));
    float hh = tanhf(xh + r * hhm);
    float hn = z * h[i] + (1.f - z) * hh;
    h[i] = hn;
    hB[i] = (__bf16)hn;
}

// ---------------------------------------------------------------------------
// Luong attention: score = h.keys, softmax over S, context = align.memory,
// and build cat = [h | context] in bf16. One block per batch element.
// ---------------------------------------------------------------------------
__global__ __launch_bounds__(256)
void k_attention(const float* __restrict__ h, const float* __restrict__ keys,
                 const float* __restrict__ memory, __bf16* __restrict__ catB) {
    int b = blockIdx.x;
    __shared__ float sc[SS];
    __shared__ float invsum;
    int tid = threadIdx.x, w = tid >> 5, lane = tid & 31;
    const float* hb = h + (size_t)b * UU;

    for (int s = w; s < SS; s += 8) {
        const float* kb = keys + ((size_t)b * SS + s) * UU;
        float acc = 0.f;
        for (int u = lane; u < UU; u += 32) acc += hb[u] * kb[u];
        #pragma unroll
        for (int off = 16; off; off >>= 1) acc += __shfl_xor(acc, off, 32);
        if (lane == 0) sc[s] = acc;
    }
    __syncthreads();
    if (tid == 0) {
        float mx = sc[0];
        for (int s = 1; s < SS; ++s) mx = fmaxf(mx, sc[s]);
        float sum = 0.f;
        for (int s = 0; s < SS; ++s) { float e = __expf(sc[s] - mx); sc[s] = e; sum += e; }
        invsum = 1.f / sum;
    }
    __syncthreads();
    float inv = invsum;
    for (int u = tid; u < UU; u += 256) {
        float acc = 0.f;
        const float* mb = memory + (size_t)b * SS * UU + u;
        #pragma unroll 4
        for (int s = 0; s < SS; ++s) acc += sc[s] * mb[(size_t)s * UU];
        catB[(size_t)b * 2 * UU + UU + u] = (__bf16)(acc * inv);
        catB[(size_t)b * 2 * UU + u]      = (__bf16)hb[u];
    }
}

// ---------------------------------------------------------------------------
// Host-side launcher
// ---------------------------------------------------------------------------
static inline size_t align256(size_t x) { return (x + 255) & ~(size_t)255; }

extern "C" void kernel_launch(void* const* d_in, const int* in_sizes, int n_in,
                              void* d_out, int out_size, void* d_ws, size_t ws_size,
                              hipStream_t stream) {
    const int*   x       = (const int*)  d_in[0];
    const float* enc     = (const float*)d_in[1];
    const float* memory  = (const float*)d_in[2];
    const float* emb     = (const float*)d_in[3];
    const float* gruK    = (const float*)d_in[4];   // [EE+UU, 3UU]
    const float* gruR    = (const float*)d_in[5];   // [UU, 3UU]
    const float* gbias   = (const float*)d_in[6];   // [2, 3UU]
    const float* memW    = (const float*)d_in[7];   // [UU, UU]
    const float* attnW   = (const float*)d_in[8];   // [2UU, UU]
    const float* fcW     = (const float*)d_in[9];   // [UU, VV]
    const float* fcB     = (const float*)d_in[10];  // [VV]
    float* out = (float*)d_out;                     // [BB, TT, VV]

    // Workspace layout (bf16 transposed weights + activations)
    char* ws = (char*)d_ws; size_t o = 0;
    __bf16* WkT  = (__bf16*)(ws + o); o = align256(o + (size_t)3*UU * (EE+UU) * 2);
    __bf16* WrT  = (__bf16*)(ws + o); o = align256(o + (size_t)3*UU * UU * 2);
    __bf16* AWT  = (__bf16*)(ws + o); o = align256(o + (size_t)UU * 2*UU * 2);
    __bf16* FWT  = (__bf16*)(ws + o); o = align256(o + (size_t)VV * UU * 2);
    __bf16* MWT  = (__bf16*)(ws + o); o = align256(o + (size_t)UU * UU * 2);
    __bf16* memB = (__bf16*)(ws + o); o = align256(o + (size_t)BB*SS*UU * 2);
    float*  keys = (float*) (ws + o); o = align256(o + (size_t)BB*SS*UU * 4);
    __bf16* cin  = (__bf16*)(ws + o); o = align256(o + (size_t)BB*(EE+UU) * 2);
    float*  hF   = (float*) (ws + o); o = align256(o + (size_t)BB*UU * 4);
    __bf16* hB   = (__bf16*)(ws + o); o = align256(o + (size_t)BB*UU * 2);
    float*  xm   = (float*) (ws + o); o = align256(o + (size_t)BB*3*UU * 4);
    float*  hm   = (float*) (ws + o); o = align256(o + (size_t)BB*3*UU * 4);
    __bf16* catB = (__bf16*)(ws + o); o = align256(o + (size_t)BB*2*UU * 2);
    float*  atF  = (float*) (ws + o); o = align256(o + (size_t)BB*UU * 4);
    __bf16* atB  = (__bf16*)(ws + o); o = align256(o + (size_t)BB*UU * 2);

    dim3 tb(32, 8);
    // Transposed bf16 weights: out[N,K] from in[K,N]
    k_transpose_f32_to_bf16<<<dim3((3*UU)/32, (EE+UU)/32), tb, 0, stream>>>(gruK,  WkT, EE+UU, 3*UU);
    k_transpose_f32_to_bf16<<<dim3((3*UU)/32, UU/32),      tb, 0, stream>>>(gruR,  WrT, UU,    3*UU);
    k_transpose_f32_to_bf16<<<dim3(UU/32, (2*UU)/32),      tb, 0, stream>>>(attnW, AWT, 2*UU,  UU);
    k_transpose_f32_to_bf16<<<dim3(VV/32, UU/32),          tb, 0, stream>>>(fcW,   FWT, UU,    VV);
    k_transpose_f32_to_bf16<<<dim3(UU/32, UU/32),          tb, 0, stream>>>(memW,  MWT, UU,    UU);

    int nmem = BB*SS*UU;
    k_f32_to_bf16<<<(nmem + 255)/256, 256, 0, stream>>>(memory, memB, nmem);
    k_init<<<(BB*UU + 255)/256, 256, 0, stream>>>(enc, hF, hB, atF);

    // keys[B*S, U] = memory @ mem_W   (M=5120, N=1024, K=1024)
    k_wmma_gemm<<<dim3(UU/128, (BB*SS)/64), 256, 0, stream>>>(memB, MWT, keys, nullptr,
                                                              BB*SS, UU, UU, UU);

    for (int t = 0; t < TT; ++t) {
        k_build_cellin<<<(BB*(EE+UU) + 255)/256, 256, 0, stream>>>(x, emb, atF, cin, t);
        // xm = cell_in @ gru_kernel       (M=64, N=3072, K=1280)
        k_wmma_gemm<<<dim3((3*UU)/128, 1), 256, 0, stream>>>(cin, WkT, xm, nullptr,
                                                             BB, 3*UU, EE+UU, 3*UU);
        // hm = h @ gru_rec_kernel         (M=64, N=3072, K=1024)
        k_wmma_gemm<<<dim3((3*UU)/128, 1), 256, 0, stream>>>(hB, WrT, hm, nullptr,
                                                             BB, 3*UU, UU, 3*UU);
        k_gru_gate<<<(BB*UU + 255)/256, 256, 0, stream>>>(xm, hm, gbias, hF, hB);
        k_attention<<<BB, 256, 0, stream>>>(hF, keys, memory, catB);
        // attn = [h|context] @ attn_W     (M=64, N=1024, K=2048)
        k_wmma_gemm<<<dim3(UU/128, 1), 256, 0, stream>>>(catB, AWT, atF, nullptr,
                                                         BB, UU, 2*UU, UU);
        k_f32_to_bf16<<<(BB*UU + 255)/256, 256, 0, stream>>>(atF, atB, BB*UU);
        // logits[t] = attn @ fc_W + fc_b  (M=64, N=32000, K=1024) -> out[:, t, :]
        k_wmma_gemm<<<dim3(VV/128, 1), 256, 0, stream>>>(atB, FWT, out + (size_t)t * VV, fcB,
                                                         BB, VV, UU, TT * VV);
    }
}
